// get_model_89721866814259
// MI455X (gfx1250) — matrix operand order repaired and checked by reading, not compiled, not run on hardware
//
#include <hip/hip_runtime.h>
#include <hip/hip_bf16.h>

// ---------------------------------------------------------------------------
// PointNet++ classification for MI455X (gfx1250, wave32, WMMA).
// All per-point MLPs run on v_wmma_f32_16x16x32_f16 (f16 in, f32 accum).
// SA1/SA2 are fused (gather -> 3 WMMA layers -> maxpool) with activations in
// LDS (320KB/WGP on CDNA5), avoiding 64-128MB global intermediates.
// ---------------------------------------------------------------------------

typedef __attribute__((ext_vector_type(16))) _Float16 v16h;
typedef __attribute__((ext_vector_type(8)))  _Float16 v8h;
typedef __attribute__((ext_vector_type(8)))  float    v8f;

#define BATCH 32
#define NPTS  4096
#define S1    512
#define NS1   32
#define R1SQ  0.04f
#define S2    128
#define NS2   64
#define R2SQ  0.16f

// ------------------------- WMMA tile helpers -------------------------------
// A operand (16x32 f16, row-major source, ld = K halves):
//   lane l holds row (l&15); 16 contiguous halves at K offset (l>=16?16:0)
__device__ __forceinline__ v16h load_a(const _Float16* __restrict__ X, int ldx,
                                       int mt, int lane, int kc) {
  const _Float16* row =
      X + (size_t)(mt * 16 + (lane & 15)) * ldx + kc * 32 + ((lane & 16) ? 16 : 0);
  union { v16h v; v8h h[2]; } u;
  u.h[0] = *(const v8h*)(row);
  u.h[1] = *(const v8h*)(row + 8);
  return u.v;
}

// B operand from W stored row-major (O x K): lane l holds output column
// (l&15); 16 contiguous halves at K offset (l>=16?16:0).
__device__ __forceinline__ v16h load_b(const _Float16* __restrict__ W, int ldw,
                                       int nt, int lane, int kc) {
  const _Float16* row =
      W + (size_t)(nt * 16 + (lane & 15)) * ldw + kc * 32 + ((lane & 16) ? 16 : 0);
  union { v16h v; v8h h[2]; } u;
  u.h[0] = *(const v8h*)(row);
  u.h[1] = *(const v8h*)(row + 8);
  return u.v;
}

__device__ __forceinline__ void wmma_tile(const _Float16* __restrict__ X, int ldx,
                                          const _Float16* __restrict__ W, int ldw,
                                          int mt, int nt, int kchunks, int lane,
                                          v8f& acc) {
  for (int kc = 0; kc < kchunks; ++kc) {
    v16h a = load_a(X, ldx, mt, lane, kc);
    v16h b = load_b(W, ldw, nt, lane, kc);
    acc = __builtin_amdgcn_wmma_f32_16x16x32_f16(
        /*neg_a=*/false, a, /*neg_b=*/false, b,
        /*c_mod=*/(short)0, acc, /*reuse_a=*/false, /*reuse_b=*/false);
  }
}

// D layout: VGPR r -> M = mt*16 + r + (lane>=16 ? 8 : 0), N = nt*16 + (lane&15)
__device__ __forceinline__ void store_tile_h(_Float16* __restrict__ Y, int ldy,
                                             int mt, int nt, int lane, const v8f& acc,
                                             const float* __restrict__ sc,
                                             const float* __restrict__ bi,
                                             int O, int relu) {
  int n = nt * 16 + (lane & 15);
  int mbase = mt * 16 + ((lane & 16) ? 8 : 0);
  float s = 1.f, bb = 0.f;
  if (n < O) { if (sc) s = sc[n]; if (bi) bb = bi[n]; }
#pragma unroll
  for (int r = 0; r < 8; ++r) {
    float v = acc[r] * s + bb;
    if (relu) v = fmaxf(v, 0.f);
    if (n >= O) v = 0.f;
    Y[(size_t)(mbase + r) * ldy + n] = (_Float16)v;
  }
}

__device__ __forceinline__ void store_tile_f(float* __restrict__ Y, int ldy,
                                             int mt, int nt, int lane, const v8f& acc,
                                             const float* __restrict__ sc,
                                             const float* __restrict__ bi,
                                             int O, int relu) {
  int n = nt * 16 + (lane & 15);
  int mbase = mt * 16 + ((lane & 16) ? 8 : 0);
  float s = 1.f, bb = 0.f;
  if (n < O) { if (sc) s = sc[n]; if (bi) bb = bi[n]; }
#pragma unroll
  for (int r = 0; r < 8; ++r) {
    float v = acc[r] * s + bb;
    if (relu) v = fmaxf(v, 0.f);
    if (n >= O) v = 0.f;
    Y[(size_t)(mbase + r) * ldy + n] = v;
  }
}

// ------------------------- prep kernels ------------------------------------

// xyz (B,6,N) -> pts (B,N,3), nrm (B,N,3)
__global__ void split_transpose_kernel(const float* __restrict__ xyz,
                                       float* __restrict__ pts,
                                       float* __restrict__ nrm) {
  int i = blockIdx.x * blockDim.x + threadIdx.x;
  if (i >= BATCH * NPTS) return;
  int b = i / NPTS, n = i % NPTS;
  const float* src = xyz + (size_t)b * 6 * NPTS + n;
  float* pp = pts + (size_t)i * 3;
  float* qq = nrm + (size_t)i * 3;
  pp[0] = src[0 * NPTS]; pp[1] = src[1 * NPTS]; pp[2] = src[2 * NPTS];
  qq[0] = src[3 * NPTS]; qq[1] = src[4 * NPTS]; qq[2] = src[5 * NPTS];
}

// W f32 (O x C) -> W f16 (Opad x Cpad) zero padded
__global__ void wconv_kernel(const float* __restrict__ W, _Float16* __restrict__ Wo,
                             int O, int C, int Opad, int Cpad) {
  int i = blockIdx.x * blockDim.x + threadIdx.x;
  if (i >= Opad * Cpad) return;
  int o = i / Cpad, c = i % Cpad;
  Wo[i] = (o < O && c < C) ? (_Float16)W[(size_t)o * C + c] : (_Float16)0.f;
}

// Farthest point sampling: one workgroup per batch, dists in registers.
__global__ void __launch_bounds__(256)
fps_kernel(const float* __restrict__ xyz, int* __restrict__ out, int N, int S) {
  __shared__ float sv[256];
  __shared__ int   si[256];
  __shared__ float c[3];
  int b = blockIdx.x, tid = threadIdx.x;
  const float* p = xyz + (size_t)b * N * 3;
  int ppt = (N + 255) / 256;               // <= 16
  float d[16];
#pragma unroll
  for (int i = 0; i < 16; ++i) d[i] = 1e10f;
  int far = 0;
  for (int s = 0; s < S; ++s) {
    if (tid == 0) {
      out[b * S + s] = far;
      c[0] = p[far * 3 + 0]; c[1] = p[far * 3 + 1]; c[2] = p[far * 3 + 2];
    }
    __syncthreads();
    float cx = c[0], cy = c[1], cz = c[2];
    float bv = -1.f; int bi = 0x7fffffff;
    for (int i = 0; i < ppt; ++i) {
      int n = tid + i * 256;
      if (n < N) {
        float dx = p[n * 3 + 0] - cx, dy = p[n * 3 + 1] - cy, dz = p[n * 3 + 2] - cz;
        float dd = dx * dx + dy * dy + dz * dz;
        d[i] = fminf(d[i], dd);
        if (d[i] > bv) { bv = d[i]; bi = n; }
      }
    }
    sv[tid] = bv; si[tid] = bi;
    __syncthreads();
    for (int off = 128; off > 0; off >>= 1) {
      if (tid < off) {
        if (sv[tid + off] > sv[tid] ||
            (sv[tid + off] == sv[tid] && si[tid + off] < si[tid])) {
          sv[tid] = sv[tid + off]; si[tid] = si[tid + off];
        }
      }
      __syncthreads();
    }
    far = si[0];
    __syncthreads();
  }
}

__global__ void gather_nx_kernel(const float* __restrict__ pts,
                                 const int* __restrict__ fidx,
                                 float* __restrict__ nx, int N, int S) {
  int i = blockIdx.x * blockDim.x + threadIdx.x;
  if (i >= BATCH * S) return;
  int b = i / S;
  const float* p = pts + ((size_t)b * N + fidx[i]) * 3;
  nx[(size_t)i * 3 + 0] = p[0];
  nx[(size_t)i * 3 + 1] = p[1];
  nx[(size_t)i * 3 + 2] = p[2];
}

// Ball query: one wave per (b,s); ballot-compacted index collection.
__global__ void ballquery_kernel(const float* __restrict__ xyz,
                                 const float* __restrict__ nx,
                                 int* __restrict__ gidx,
                                 int N, int S, int ns, float r2) {
  int wid = (blockIdx.x * blockDim.x + threadIdx.x) >> 5;
  int lane = threadIdx.x & 31;
  if (wid >= BATCH * S) return;
  int b = wid / S;
  const float* p = xyz + (size_t)b * N * 3;
  float cx = nx[(size_t)wid * 3 + 0];
  float cy = nx[(size_t)wid * 3 + 1];
  float cz = nx[(size_t)wid * 3 + 2];
  int* out = gidx + (size_t)wid * ns;
  int count = 0, first = -1;
  for (int base = 0; base < N && count < ns; base += 32) {
    int n = base + lane;
    bool pred = false;
    if (n < N) {
      float dx = p[n * 3 + 0] - cx, dy = p[n * 3 + 1] - cy, dz = p[n * 3 + 2] - cz;
      pred = (dx * dx + dy * dy + dz * dz) <= r2;
    }
    unsigned mask = (unsigned)__ballot(pred);
    if (mask) {
      if (first < 0) first = base + __ffs(mask) - 1;
      int pos = count + __popc(mask & ((1u << lane) - 1u));
      if (pred && pos < ns) out[pos] = n;
      count += __popc(mask);
    }
  }
  if (first < 0) first = 0;
  for (int pos = count + lane; pos < ns; pos += 32) out[pos] = first;
}

// ---------------- fused set-abstraction: gather + 3 WMMA layers + maxpool ---
// One workgroup (8 waves) per (b,s) group; activations in LDS.
__global__ void __launch_bounds__(256)
sa_fused_kernel(const float* __restrict__ xyz,      // (B,N,3) source points
                const float* __restrict__ nx,       // (B,S,3) centers
                const int* __restrict__ gidx,       // (B,S,ns)
                const _Float16* __restrict__ pfeat, int Cf,   // f16 point feats
                const float* __restrict__ pfeat32, int Cf32,  // f32 point feats
                int N, int S, int ns, int K0,
                const _Float16* __restrict__ W1, const float* __restrict__ s1,
                const float* __restrict__ b1, int O1,
                const _Float16* __restrict__ W2, const float* __restrict__ s2,
                const float* __restrict__ b2, int O2,
                const _Float16* __restrict__ W3, const float* __restrict__ s3,
                const float* __restrict__ b3, int O3,
                _Float16* __restrict__ outp)        // (B,S,O3)
{
  extern __shared__ _Float16 smem[];
  _Float16* X0 = smem;
  _Float16* A1 = X0 + ns * K0;
  _Float16* A2 = A1 + ns * O1;
  _Float16* A3 = A2 + ns * O2;

  int g = blockIdx.x;                 // g = b*S + s
  int b = g / S;
  int tid = threadIdx.x;
  const int* gi = gidx + (size_t)g * ns;

  // gather features into LDS (zero padded to K0)
  for (int e = tid; e < ns * K0; e += blockDim.x) {
    int r = e / K0, c = e % K0;
    int k = gi[r];
    float v = 0.f;
    if (c < 3) {
      v = xyz[((size_t)b * N + k) * 3 + c] - nx[(size_t)g * 3 + c];
    } else if (pfeat && c < 3 + Cf) {
      v = (float)pfeat[((size_t)b * N + k) * Cf + (c - 3)];
    } else if (pfeat32 && c < 3 + Cf32) {
      v = pfeat32[((size_t)b * N + k) * Cf32 + (c - 3)];
    }
    X0[e] = (_Float16)v;
  }
  __syncthreads();

  int lane = tid & 31, wave = tid >> 5, nw = blockDim.x >> 5;
  int mtiles = ns / 16;

  // layer 1
  for (int t = wave; t < mtiles * (O1 / 16); t += nw) {
    int mt = t / (O1 / 16), nt = t % (O1 / 16);
    v8f acc = {0.f, 0.f, 0.f, 0.f, 0.f, 0.f, 0.f, 0.f};
    wmma_tile(X0, K0, W1, K0, mt, nt, K0 / 32, lane, acc);
    store_tile_h(A1, O1, mt, nt, lane, acc, s1, b1, O1, 1);
  }
  __syncthreads();
  // layer 2
  for (int t = wave; t < mtiles * (O2 / 16); t += nw) {
    int mt = t / (O2 / 16), nt = t % (O2 / 16);
    v8f acc = {0.f, 0.f, 0.f, 0.f, 0.f, 0.f, 0.f, 0.f};
    wmma_tile(A1, O1, W2, O1, mt, nt, O1 / 32, lane, acc);
    store_tile_h(A2, O2, mt, nt, lane, acc, s2, b2, O2, 1);
  }
  __syncthreads();
  // layer 3
  for (int t = wave; t < mtiles * (O3 / 16); t += nw) {
    int mt = t / (O3 / 16), nt = t % (O3 / 16);
    v8f acc = {0.f, 0.f, 0.f, 0.f, 0.f, 0.f, 0.f, 0.f};
    wmma_tile(A2, O2, W3, O2, mt, nt, O2 / 32, lane, acc);
    store_tile_h(A3, O3, mt, nt, lane, acc, s3, b3, O3, 1);
  }
  __syncthreads();
  // maxpool over ns samples
  for (int o = tid; o < O3; o += blockDim.x) {
    float m = -1e30f;
    for (int r = 0; r < ns; ++r) m = fmaxf(m, (float)A3[r * O3 + o]);
    outp[(size_t)g * O3 + o] = (_Float16)m;
  }
}

// ---------------- generic WMMA MLP layer over global rows -------------------
__global__ void __launch_bounds__(256)
mlp_wmma_kernel(const _Float16* __restrict__ X,  // R x K
                const _Float16* __restrict__ W,  // Opad x K
                const float* __restrict__ sc, const float* __restrict__ bi,
                _Float16* __restrict__ Yh, float* __restrict__ Yf,
                int Rtiles, int K, int Opad, int O, int relu) {
  int tile = blockIdx.x * (blockDim.x >> 5) + (threadIdx.x >> 5);
  int ntiles = Opad / 16;
  if (tile >= Rtiles * ntiles) return;
  int mt = tile / ntiles, nt = tile % ntiles;
  int lane = threadIdx.x & 31;
  v8f acc = {0.f, 0.f, 0.f, 0.f, 0.f, 0.f, 0.f, 0.f};
  wmma_tile(X, K, W, K, mt, nt, K / 32, lane, acc);
  if (Yh) store_tile_h(Yh, Opad, mt, nt, lane, acc, sc, bi, O, relu);
  if (Yf) store_tile_f(Yf, Opad, mt, nt, lane, acc, sc, bi, O, relu);
}

// SA3 feature build: (B,128) rows of [l2_xyz(3) | l2_p(256) | pad] -> f16
__global__ void build_feat3_kernel(const float* __restrict__ nx2,
                                   const _Float16* __restrict__ l2p,
                                   _Float16* __restrict__ F3, int C2, int K0) {
  int i = blockIdx.x * blockDim.x + threadIdx.x;
  int total = BATCH * S2 * K0;
  if (i >= total) return;
  int c = i % K0, rs = i / K0;
  float v = 0.f;
  if (c < 3)            v = nx2[(size_t)rs * 3 + c];
  else if (c < 3 + C2)  v = (float)l2p[(size_t)rs * C2 + (c - 3)];
  F3[i] = (_Float16)v;
}

__global__ void maxpool_kernel(const _Float16* __restrict__ Y,
                               _Float16* __restrict__ out,
                               int groups, int ns, int O) {
  int i = blockIdx.x * blockDim.x + threadIdx.x;
  if (i >= groups * O) return;
  int g = i / O, o = i % O;
  const _Float16* p = Y + (size_t)g * ns * O + o;
  float m = -1e30f;
  for (int r = 0; r < ns; ++r) m = fmaxf(m, (float)p[(size_t)r * O]);
  out[(size_t)g * O + o] = (_Float16)m;
}

__global__ void logsoftmax_kernel(const float* __restrict__ logits,
                                  float* __restrict__ out, int C, int Cpad) {
  int b = blockIdx.x * blockDim.x + threadIdx.x;
  if (b >= BATCH) return;
  const float* x = logits + (size_t)b * Cpad;
  float mx = -1e30f;
  for (int c = 0; c < C; ++c) mx = fmaxf(mx, x[c]);
  float s = 0.f;
  for (int c = 0; c < C; ++c) s += expf(x[c] - mx);
  float ls = logf(s);
  for (int c = 0; c < C; ++c) out[(size_t)b * C + c] = x[c] - mx - ls;
}

// ---------------------------------------------------------------------------
extern "C" void kernel_launch(void* const* d_in, const int* in_sizes, int n_in,
                              void* d_out, int out_size, void* d_ws, size_t ws_size,
                              hipStream_t stream) {
  (void)in_sizes; (void)n_in; (void)out_size; (void)ws_size;

  // ---- inputs (setup_inputs insertion order) ----
  // 0: xyz ; sa1 L0..L2 (W,s,b) = 1..9 ; sa2 = 10..18 ; sa3 = 19..27 ;
  // fc1 (W,s,b) = 28..30 ; conv1 (W,b) = 31..32
  const float* xyz = (const float*)d_in[0];
  const float *Wsa[3][3], *Ssa[3][3], *Bsa[3][3];
  for (int st = 0; st < 3; ++st)
    for (int l = 0; l < 3; ++l) {
      int base = 1 + st * 9 + l * 3;
      Wsa[st][l] = (const float*)d_in[base + 0];
      Ssa[st][l] = (const float*)d_in[base + 1];
      Bsa[st][l] = (const float*)d_in[base + 2];
    }
  const float* Wfc = (const float*)d_in[28];
  const float* Sfc = (const float*)d_in[29];
  const float* Bfc = (const float*)d_in[30];
  const float* Wcv = (const float*)d_in[31];
  const float* Bcv = (const float*)d_in[32];

  // ---- workspace allocator ----
  char* ws = (char*)d_ws;
  size_t off = 0;
  auto alloc = [&](size_t bytes) -> void* {
    off = (off + 255) & ~(size_t)255;
    void* p = ws + off;
    off += bytes;
    return p;
  };

  float* pts  = (float*)alloc((size_t)BATCH * NPTS * 3 * 4);
  float* nrm  = (float*)alloc((size_t)BATCH * NPTS * 3 * 4);
  int*   fidx1 = (int*)alloc((size_t)BATCH * S1 * 4);
  int*   fidx2 = (int*)alloc((size_t)BATCH * S2 * 4);
  float* nx1  = (float*)alloc((size_t)BATCH * S1 * 3 * 4);
  float* nx2  = (float*)alloc((size_t)BATCH * S2 * 3 * 4);
  int*   gidx1 = (int*)alloc((size_t)BATCH * S1 * NS1 * 4);
  int*   gidx2 = (int*)alloc((size_t)BATCH * S2 * NS2 * 4);
  _Float16* l1p = (_Float16*)alloc((size_t)BATCH * S1 * 128 * 2);
  _Float16* l2p = (_Float16*)alloc((size_t)BATCH * S2 * 256 * 2);
  _Float16* F3  = (_Float16*)alloc((size_t)BATCH * S2 * 288 * 2);
  _Float16* Y3a = (_Float16*)alloc((size_t)BATCH * S2 * 256 * 2);
  _Float16* Y3b = (_Float16*)alloc((size_t)BATCH * S2 * 512 * 2);
  _Float16* Y3c = (_Float16*)alloc((size_t)BATCH * S2 * 1024 * 2);
  _Float16* gfe = (_Float16*)alloc((size_t)BATCH * 1024 * 2);
  _Float16* fco = (_Float16*)alloc((size_t)BATCH * 512 * 2);
  float*    lgt = (float*)alloc((size_t)BATCH * 48 * 4);

  // packed f16 weights (Opad x Kpad)
  struct WSpec { const float* W; int O, C, Opad, Cpad; _Float16* out; };
  _Float16* w1[3]; _Float16* w2[3]; _Float16* w3[3]; _Float16 *wfc, *wcv;
  w1[0] = (_Float16*)alloc(64 * 32 * 2);
  w1[1] = (_Float16*)alloc(64 * 64 * 2);
  w1[2] = (_Float16*)alloc(128 * 64 * 2);
  w2[0] = (_Float16*)alloc(128 * 160 * 2);
  w2[1] = (_Float16*)alloc(128 * 128 * 2);
  w2[2] = (_Float16*)alloc(256 * 128 * 2);
  w3[0] = (_Float16*)alloc(256 * 288 * 2);
  w3[1] = (_Float16*)alloc(512 * 256 * 2);
  w3[2] = (_Float16*)alloc(1024 * 512 * 2);
  wfc   = (_Float16*)alloc(512 * 1024 * 2);
  wcv   = (_Float16*)alloc(48 * 512 * 2);

  WSpec specs[11] = {
    {Wsa[0][0],  64,   6,  64,  32, w1[0]}, {Wsa[0][1],  64,  64,  64,  64, w1[1]},
    {Wsa[0][2], 128,  64, 128,  64, w1[2]}, {Wsa[1][0], 128, 131, 128, 160, w2[0]},
    {Wsa[1][1], 128, 128, 128, 128, w2[1]}, {Wsa[1][2], 256, 128, 256, 128, w2[2]},
    {Wsa[2][0], 256, 259, 256, 288, w3[0]}, {Wsa[2][1], 512, 256, 512, 256, w3[1]},
    {Wsa[2][2],1024, 512,1024, 512, w3[2]}, {Wfc, 512, 1024, 512, 1024, wfc},
    {Wcv, 40, 512, 48, 512, wcv}};
  for (int i = 0; i < 11; ++i) {
    int elems = specs[i].Opad * specs[i].Cpad;
    wconv_kernel<<<(elems + 255) / 256, 256, 0, stream>>>(
        specs[i].W, specs[i].out, specs[i].O, specs[i].C, specs[i].Opad,
        specs[i].Cpad);
  }

  // ---- stage 0: transpose/split ----
  split_transpose_kernel<<<(BATCH * NPTS + 255) / 256, 256, 0, stream>>>(xyz, pts, nrm);

  // ---- SA1 ----
  fps_kernel<<<BATCH, 256, 0, stream>>>(pts, fidx1, NPTS, S1);
  gather_nx_kernel<<<(BATCH * S1 + 255) / 256, 256, 0, stream>>>(pts, fidx1, nx1, NPTS, S1);
  ballquery_kernel<<<(BATCH * S1 * 32 + 255) / 256, 256, 0, stream>>>(
      pts, nx1, gidx1, NPTS, S1, NS1, R1SQ);
  {
    size_t lds = (size_t)(NS1 * 32 + NS1 * 64 + NS1 * 64 + NS1 * 128) * 2;  // 18KB
    sa_fused_kernel<<<BATCH * S1, 256, lds, stream>>>(
        pts, nx1, gidx1, nullptr, 0, nrm, 3, NPTS, S1, NS1, 32,
        w1[0], Ssa[0][0], Bsa[0][0], 64,
        w1[1], Ssa[0][1], Bsa[0][1], 64,
        w1[2], Ssa[0][2], Bsa[0][2], 128, l1p);
  }

  // ---- SA2 ----
  fps_kernel<<<BATCH, 256, 0, stream>>>(nx1, fidx2, S1, S2);
  gather_nx_kernel<<<(BATCH * S2 + 255) / 256, 256, 0, stream>>>(nx1, fidx2, nx2, S1, S2);
  ballquery_kernel<<<(BATCH * S2 * 32 + 255) / 256, 256, 0, stream>>>(
      nx1, nx2, gidx2, S1, S2, NS2, R2SQ);
  {
    size_t lds = (size_t)(NS2 * 160 + NS2 * 128 + NS2 * 128 + NS2 * 256) * 2;  // 84KB
    sa_fused_kernel<<<BATCH * S2, 256, lds, stream>>>(
        nx1, nx2, gidx2, l1p, 128, nullptr, 0, S1, S2, NS2, 160,
        w2[0], Ssa[1][0], Bsa[1][0], 128,
        w2[1], Ssa[1][1], Bsa[1][1], 128,
        w2[2], Ssa[1][2], Bsa[1][2], 256, l2p);
  }

  // ---- SA3 (global, group_all): rows = B*128 ----
  build_feat3_kernel<<<(BATCH * S2 * 288 + 255) / 256, 256, 0, stream>>>(
      nx2, l2p, F3, 256, 288);
  {
    int Rt = (BATCH * S2) / 16;  // 256 row tiles
    int t1 = Rt * (256 / 16), t2 = Rt * (512 / 16), t3 = Rt * (1024 / 16);
    mlp_wmma_kernel<<<(t1 + 7) / 8, 256, 0, stream>>>(
        F3, w3[0], Ssa[2][0], Bsa[2][0], Y3a, nullptr, Rt, 288, 256, 256, 1);
    mlp_wmma_kernel<<<(t2 + 7) / 8, 256, 0, stream>>>(
        Y3a, w3[1], Ssa[2][1], Bsa[2][1], Y3b, nullptr, Rt, 256, 512, 512, 1);
    mlp_wmma_kernel<<<(t3 + 7) / 8, 256, 0, stream>>>(
        Y3b, w3[2], Ssa[2][2], Bsa[2][2], Y3c, nullptr, Rt, 512, 1024, 1024, 1);
  }
  maxpool_kernel<<<(BATCH * 1024 + 255) / 256, 256, 0, stream>>>(Y3c, gfe, BATCH, S2, 1024);

  // ---- head ----
  {
    int Rt = BATCH / 16;  // 2
    int tfc = Rt * (512 / 16), tcv = Rt * (48 / 16);
    mlp_wmma_kernel<<<(tfc + 7) / 8, 256, 0, stream>>>(
        gfe, wfc, Sfc, Bfc, fco, nullptr, Rt, 1024, 512, 512, 1);
    mlp_wmma_kernel<<<(tcv + 7) / 8, 256, 0, stream>>>(
        fco, wcv, nullptr, Bcv, nullptr, lgt, Rt, 512, 48, 40, 0);
  }
  logsoftmax_kernel<<<1, 32, 0, stream>>>(lgt, (float*)d_out, 40, 48);
}
